// AttentionBlock_50440095924455
// MI455X (gfx1250) — matrix operand order repaired and checked by reading, not compiled, not run on hardware
//
#include <hip/hip_runtime.h>
#include <hip/hip_bf16.h>

#define B_      4
#define C_      256
#define NTOK    4096
#define GROUPS  32
#define CPG     8           // channels per group
#define HEADS   4
#define BH      16          // B_*HEADS
#define HD      64          // head dim
#define QK_SCALE 0.125f     // HD^-0.5 (both 0.25-power factors folded into q)

typedef __bf16 bf16;
typedef __attribute__((ext_vector_type(16))) __bf16 v16bf;
typedef __attribute__((ext_vector_type(8)))  __bf16 v8bf;
typedef __attribute__((ext_vector_type(8)))  float   v8f;

// ---- WMMA fragment loaders (wave32, per ISA 7.12.2 layouts) ----
// A (16x32 bf16): lane m=l%16, half=l/16 holds K in [8h,8h+8) and [16+8h,16+8h+8)
static __device__ inline v16bf frag_a_bf16(const bf16* rowp) {
  const int half = (threadIdx.x >> 4) & 1;
  v8bf lo = *(const v8bf*)(rowp + 8 * half);
  v8bf hi = *(const v8bf*)(rowp + 16 + 8 * half);
  return __builtin_shufflevector(lo, hi, 0,1,2,3,4,5,6,7,8,9,10,11,12,13,14,15);
}
// B (32x16 bf16) from B^T[N][K]: lane n=l%16, half=l/16 holds K in [8h,8h+8) and [16+8h,16+8h+8)
static __device__ inline v16bf frag_b_bf16(const bf16* rowp) {
  const int half = (threadIdx.x >> 4) & 1;
  v8bf lo = *(const v8bf*)(rowp + 8 * half);
  v8bf hi = *(const v8bf*)(rowp + 16 + 8 * half);
  return __builtin_shufflevector(lo, hi, 0,1,2,3,4,5,6,7,8,9,10,11,12,13,14,15);
}
// A fragment from f32 LDS row (softmax probabilities), converted to bf16
static __device__ inline v16bf frag_a_f32(const float* rowp) {
  const int half = (threadIdx.x >> 4) & 1;
  const float* p0 = rowp + 8 * half;
  const float* p1 = rowp + 16 + 8 * half;
  v16bf r;
#pragma unroll
  for (int i = 0; i < 8; ++i) { r[i] = (__bf16)p0[i]; r[8 + i] = (__bf16)p1[i]; }
  return r;
}

// ---- Kernel 1: GroupNorm -> xnT [B][t][C] in bf16 ----
__global__ void gn_kernel(const float* __restrict__ x, const float* __restrict__ gw,
                          const float* __restrict__ gb, bf16* __restrict__ xnT) {
  __shared__ float rs[256], rq[256];
  const int b = blockIdx.x / GROUPS;
  const int g = blockIdx.x % GROUPS;
  const int c0 = g * CPG;
  const float* xb = x + (size_t)b * C_ * NTOK;
  float s = 0.f, q = 0.f;
  for (int idx = threadIdx.x; idx < CPG * NTOK; idx += 256) {
    float v = xb[(c0 + (idx >> 12)) * NTOK + (idx & (NTOK - 1))];
    s += v; q += v * v;
  }
  rs[threadIdx.x] = s; rq[threadIdx.x] = q;
  __syncthreads();
  for (int off = 128; off > 0; off >>= 1) {
    if ((int)threadIdx.x < off) {
      rs[threadIdx.x] += rs[threadIdx.x + off];
      rq[threadIdx.x] += rq[threadIdx.x + off];
    }
    __syncthreads();
  }
  const float inv_n = 1.0f / (float)(CPG * NTOK);
  const float mu   = rs[0] * inv_n;
  const float var  = rq[0] * inv_n - mu * mu;
  const float rsig = rsqrtf(var + 1e-5f);
  for (int idx = threadIdx.x; idx < CPG * NTOK; idx += 256) {
    int c = c0 + (idx >> 12);
    int t = idx & (NTOK - 1);
    float v = (xb[c * NTOK + t] - mu) * rsig * gw[c] + gb[c];
    xnT[((size_t)b * NTOK + t) * C_ + c] = (bf16)v;
  }
}

// ---- Kernel 2: convert weights to bf16 ----
__global__ void wconv_kernel(const float* __restrict__ qkvw, const float* __restrict__ projw,
                             bf16* __restrict__ qkvw_bf, bf16* __restrict__ projw_bf) {
  int i = blockIdx.x * 256 + threadIdx.x;
  if (i < 3 * C_ * C_) qkvw_bf[i] = (bf16)qkvw[i];
  if (i < C_ * C_)     projw_bf[i] = (bf16)projw[i];
}

// ---- Kernel 3: QKV GEMM (per b: [768x256] x [256x4096]) + head-split scatter ----
// Wave computes a 16x64 strip (4 N-tiles) so A fragments are loaded once per K-step.
__global__ void qkv_kernel(const bf16* __restrict__ w, const float* __restrict__ bias,
                           const bf16* __restrict__ xnT, bf16* __restrict__ qT,
                           bf16* __restrict__ kT, bf16* __restrict__ vmat) {
  const int wid = blockIdx.x * 8 + (threadIdx.x >> 5);   // 4 * 48 * 64 waves
  const int b  = wid / (48 * 64);
  const int r0 = wid % (48 * 64);
  const int mt = r0 / 64;
  const int n4 = r0 % 64;                                // strip of 4 N-tiles
  const int lane = threadIdx.x & 31;
  const int half = lane >> 4, col = lane & 15;
  const bf16* arow = w + (size_t)(mt * 16 + col) * C_;
  const bf16* brow0 = xnT + ((size_t)b * NTOK + (n4 * 4 + 0) * 16 + col) * C_;
  const bf16* brow1 = xnT + ((size_t)b * NTOK + (n4 * 4 + 1) * 16 + col) * C_;
  const bf16* brow2 = xnT + ((size_t)b * NTOK + (n4 * 4 + 2) * 16 + col) * C_;
  const bf16* brow3 = xnT + ((size_t)b * NTOK + (n4 * 4 + 3) * 16 + col) * C_;
  v8f acc0 = {}, acc1 = {}, acc2 = {}, acc3 = {};
#pragma unroll
  for (int kk = 0; kk < C_; kk += 32) {
    v16bf a = frag_a_bf16(arow + kk);
    acc0 = __builtin_amdgcn_wmma_f32_16x16x32_bf16(false, a, false, frag_b_bf16(brow0 + kk), (short)0, acc0, false, false);
    acc1 = __builtin_amdgcn_wmma_f32_16x16x32_bf16(false, a, false, frag_b_bf16(brow1 + kk), (short)0, acc1, false, false);
    acc2 = __builtin_amdgcn_wmma_f32_16x16x32_bf16(false, a, false, frag_b_bf16(brow2 + kk), (short)0, acc2, false, false);
    acc3 = __builtin_amdgcn_wmma_f32_16x16x32_bf16(false, a, false, frag_b_bf16(brow3 + kk), (short)0, acc3, false, false);
  }
  const int o0    = mt * 16;
  const int head  = o0 / 192;           // tile-uniform (16 | 192)
  const int rr    = o0 % 192;
  const int piece = rr / 64;            // 0=q, 1=k, 2=v (tile-uniform)
  const int row0  = rr % 64;
  const int hb    = b * HEADS + head;
  v8f* accs[4] = { &acc0, &acc1, &acc2, &acc3 };
#pragma unroll
  for (int i = 0; i < 4; ++i) {
    const int t = (n4 * 4 + i) * 16 + col;
#pragma unroll
    for (int j = 0; j < 8; ++j) {
      const int ol = j + 8 * half;      // D-tile row (M) per ISA layout
      float val = (*accs[i])[j] + bias[o0 + ol];
      const int row = row0 + ol;
      if (piece == 0)      qT[((size_t)hb * NTOK + t) * HD + row] = (bf16)(val * QK_SCALE);
      else if (piece == 1) kT[((size_t)hb * NTOK + t) * HD + row] = (bf16)val;
      else                 vmat[((size_t)hb * HD + row) * NTOK + t] = (bf16)val;
    }
  }
}

// ---- Kernel 4: attention; block = (head-batch, 16 queries). 256-KB score tile in LDS. ----
__global__ void __launch_bounds__(256, 1)
attn_kernel(const bf16* __restrict__ qT, const bf16* __restrict__ kT,
            const bf16* __restrict__ vmat, bf16* __restrict__ hT) {
  __shared__ float sc[16 * NTOK];       // 256 KB of the 320 KB WGP LDS
  __shared__ float red[256];
  __shared__ float rowmax[16], rowinv[16];
  __shared__ float acc2buf[4 * 256];

  const int hb   = blockIdx.x / (NTOK / 16);
  const int tq0  = (blockIdx.x % (NTOK / 16)) * 16;
  const int lane = threadIdx.x & 31;
  const int wid  = threadIdx.x >> 5;
  const int half = lane >> 4, col = lane & 15;

  // Stage 1: scores S[16][4096] = (q^T)(k); q A-fragments hoisted (loop-invariant)
  const bf16* qrow = qT + ((size_t)hb * NTOK + tq0 + col) * HD;
  const v16bf qa0 = frag_a_bf16(qrow);
  const v16bf qa1 = frag_a_bf16(qrow + 32);
  const bf16* kbase = kT + ((size_t)hb * NTOK + col) * HD;
  for (int nt = wid * 32; nt < (wid + 1) * 32; ++nt) {
    const bf16* krow = kbase + (size_t)(nt * 16) * HD;
    __builtin_prefetch(krow + 16 * HD, 0, 0);   // stage next k-tile (global_prefetch_b8)
    v8f acc = {};
    acc = __builtin_amdgcn_wmma_f32_16x16x32_bf16(false, qa0, false, frag_b_bf16(krow), (short)0, acc, false, false);
    acc = __builtin_amdgcn_wmma_f32_16x16x32_bf16(false, qa1, false, frag_b_bf16(krow + 32), (short)0, acc, false, false);
#pragma unroll
    for (int j = 0; j < 8; ++j)
      sc[(j + 8 * half) * NTOK + nt * 16 + col] = acc[j];
  }
  __syncthreads();

  // Exact softmax over each of the 16 rows, fully in LDS
  const int srow = threadIdx.x >> 4;
  const int scol = threadIdx.x & 15;
  float m = -3.0e38f;
  for (int i = scol; i < NTOK; i += 16) m = fmaxf(m, sc[srow * NTOK + i]);
  red[threadIdx.x] = m;
  __syncthreads();
  if (scol == 0) {
    float mm = red[srow << 4];
    for (int j = 1; j < 16; ++j) mm = fmaxf(mm, red[(srow << 4) + j]);
    rowmax[srow] = mm;
  }
  __syncthreads();
  const float rm = rowmax[srow];
  float ssum = 0.f;
  for (int i = scol; i < NTOK; i += 16) ssum += __expf(sc[srow * NTOK + i] - rm);
  red[threadIdx.x] = ssum;
  __syncthreads();
  if (scol == 0) {
    float ss = 0.f;
    for (int j = 0; j < 16; ++j) ss += red[(srow << 4) + j];
    rowinv[srow] = 1.0f / ss;
  }
  __syncthreads();
  for (int idx = threadIdx.x; idx < 16 * NTOK; idx += 256) {
    int r = idx >> 12;
    sc[idx] = __expf(sc[idx] - rowmax[r]) * rowinv[r];
  }
  __syncthreads();

  // Stage 2: hT_tile[16 t][64 c] = P[16][4096] x v^T[4096][64]; K split over 2 wave groups
  const int ct = wid & 3;
  const int khalf = wid >> 2;
  const float* arow = sc + col * NTOK;
  const bf16* brow = vmat + ((size_t)hb * HD + ct * 16 + col) * NTOK;
  v8f acc = {};
#pragma unroll 2
  for (int kk = khalf * (NTOK / 2); kk < (khalf + 1) * (NTOK / 2); kk += 32) {
    v16bf a = frag_a_f32(arow + kk);
    v16bf b = frag_b_bf16(brow + kk);
    acc = __builtin_amdgcn_wmma_f32_16x16x32_bf16(false, a, false, b, (short)0, acc, false, false);
  }
  if (khalf == 1) {
#pragma unroll
    for (int j = 0; j < 8; ++j) acc2buf[ct * 256 + j * 32 + lane] = acc[j];
  }
  __syncthreads();
  if (khalf == 0) {
    const int bb = hb / HEADS, head = hb % HEADS;
#pragma unroll
    for (int j = 0; j < 8; ++j) {
      float val = acc[j] + acc2buf[ct * 256 + j * 32 + lane];
      int t  = tq0 + j + 8 * half;
      int cg = head * HD + ct * 16 + col;
      hT[((size_t)bb * NTOK + t) * C_ + cg] = (bf16)val;
    }
  }
}

// ---- Kernel 5: proj GEMM + bias + residual (fp32 out); 16x64 strip per wave ----
__global__ void proj_kernel(const bf16* __restrict__ w, const float* __restrict__ bias,
                            const bf16* __restrict__ hT, const float* __restrict__ x,
                            float* __restrict__ out) {
  const int wid = blockIdx.x * 8 + (threadIdx.x >> 5);   // 4 * 16 * 64 waves
  const int b  = wid / (16 * 64);
  const int r0 = wid % (16 * 64);
  const int mt = r0 / 64;
  const int n4 = r0 % 64;
  const int lane = threadIdx.x & 31;
  const int half = lane >> 4, col = lane & 15;
  const bf16* arow = w + (size_t)(mt * 16 + col) * C_;
  const bf16* brow0 = hT + ((size_t)b * NTOK + (n4 * 4 + 0) * 16 + col) * C_;
  const bf16* brow1 = hT + ((size_t)b * NTOK + (n4 * 4 + 1) * 16 + col) * C_;
  const bf16* brow2 = hT + ((size_t)b * NTOK + (n4 * 4 + 2) * 16 + col) * C_;
  const bf16* brow3 = hT + ((size_t)b * NTOK + (n4 * 4 + 3) * 16 + col) * C_;
  v8f acc0 = {}, acc1 = {}, acc2 = {}, acc3 = {};
#pragma unroll
  for (int kk = 0; kk < C_; kk += 32) {
    v16bf a = frag_a_bf16(arow + kk);
    acc0 = __builtin_amdgcn_wmma_f32_16x16x32_bf16(false, a, false, frag_b_bf16(brow0 + kk), (short)0, acc0, false, false);
    acc1 = __builtin_amdgcn_wmma_f32_16x16x32_bf16(false, a, false, frag_b_bf16(brow1 + kk), (short)0, acc1, false, false);
    acc2 = __builtin_amdgcn_wmma_f32_16x16x32_bf16(false, a, false, frag_b_bf16(brow2 + kk), (short)0, acc2, false, false);
    acc3 = __builtin_amdgcn_wmma_f32_16x16x32_bf16(false, a, false, frag_b_bf16(brow3 + kk), (short)0, acc3, false, false);
  }
  v8f* accs[4] = { &acc0, &acc1, &acc2, &acc3 };
#pragma unroll
  for (int i = 0; i < 4; ++i) {
    const int t = (n4 * 4 + i) * 16 + col;
#pragma unroll
    for (int j = 0; j < 8; ++j) {
      const int o = mt * 16 + j + 8 * half;
      const size_t idx = ((size_t)b * C_ + o) * NTOK + t;
      out[idx] = x[idx] + bias[o] + (*accs[i])[j];
    }
  }
}

extern "C" void kernel_launch(void* const* d_in, const int* in_sizes, int n_in,
                              void* d_out, int out_size, void* d_ws, size_t ws_size,
                              hipStream_t stream) {
  (void)in_sizes; (void)n_in; (void)out_size; (void)ws_size;
  const float* x      = (const float*)d_in[0];
  const float* gn_w   = (const float*)d_in[1];
  const float* gn_b   = (const float*)d_in[2];
  const float* qkv_w  = (const float*)d_in[3];
  const float* qkv_b  = (const float*)d_in[4];
  const float* proj_w = (const float*)d_in[5];
  const float* proj_b = (const float*)d_in[6];
  float* out = (float*)d_out;

  char* ws = (char*)d_ws;
  bf16* xnT      = (bf16*)(ws + 0);                 //  8 MB  [B][t][C]
  bf16* qT       = (bf16*)(ws + 8388608);           //  8 MB  [BH][t][HD]
  bf16* kT       = (bf16*)(ws + 16777216);          //  8 MB  [BH][t][HD]
  bf16* vmat     = (bf16*)(ws + 25165824);          //  8 MB  [BH][HD][t]
  bf16* hT       = (bf16*)(ws + 33554432);          //  8 MB  [B][t][C]
  bf16* qkvw_bf  = (bf16*)(ws + 41943040);          //  384 KB
  bf16* projw_bf = (bf16*)(ws + 42336256);          //  128 KB

  gn_kernel<<<B_ * GROUPS, 256, 0, stream>>>(x, gn_w, gn_b, xnT);
  wconv_kernel<<<(3 * C_ * C_ + 255) / 256, 256, 0, stream>>>(qkv_w, proj_w, qkvw_bf, projw_bf);
  qkv_kernel<<<(B_ * 48 * 64) / 8, 256, 0, stream>>>(qkvw_bf, qkv_b, xnT, qT, kT, vmat);
  attn_kernel<<<BH * (NTOK / 16), 256, 0, stream>>>(qT, kT, vmat, hT);
  proj_kernel<<<(B_ * 16 * 64) / 8, 256, 0, stream>>>(projw_bf, proj_b, hT, x, out);
}
